// OptimizedFP4Linear_42649025249834
// MI455X (gfx1250) — compile-verified, matrix operand order checked by reading, not run on hardware
//
#include <hip/hip_runtime.h>

typedef _Float16 v16h __attribute__((ext_vector_type(16)));
typedef _Float16 v8h  __attribute__((ext_vector_type(8)));
typedef float    v8f  __attribute__((ext_vector_type(8)));

#define IN_F   4096
#define OUT_F  4096
#define BM     128
#define BN     128
#define BK     32
#define SLDS   40   // padded LDS row stride in halves (80B)

// ---------------------------------------------------------------------------
// Pass 1: FP4 (E2M1 + sign) dequant -> f16 weight matrix W[OUT_F][IN_F] in ws.
// packed[j] holds one byte: element 2j = high nibble, 2j+1 = low nibble.
// Scale block = element/128 = j/64 (both nibbles share a block).
// ---------------------------------------------------------------------------
__global__ __launch_bounds__(256) void dequant_fp4_kernel(
    const int* __restrict__ packed, const float* __restrict__ scales,
    _Float16* __restrict__ W, int n_packed)
{
    int j = blockIdx.x * 256 + threadIdx.x;
    if (j >= n_packed) return;

    unsigned b = ((unsigned)packed[j]) & 0xFFu;
    float s = scales[j >> 6];

    const float lut[8] = {0.0f, 0.5f, 1.0f, 1.5f, 2.0f, 3.0f, 4.0f, 6.0f};
    unsigned hi = (b >> 4) & 15u;
    unsigned lo = b & 15u;
    float v0 = lut[hi & 7u]; if (hi & 8u) v0 = -v0;
    float v1 = lut[lo & 7u]; if (lo & 8u) v1 = -v1;
    v0 *= s; v1 *= s;

    union { _Float16 h[2]; unsigned u; } pk;
    pk.h[0] = (_Float16)v0;
    pk.h[1] = (_Float16)v1;
    ((unsigned*)W)[j] = pk.u;
}

// ---------------------------------------------------------------------------
// Pass 2: out[M,N] = X[M,K] * W[N,K]^T + bias, f16 WMMA, f32 accumulate.
// 128 threads = 4 wave32 (2x2). 128x128 block tile, 64x64 wave tile (4x4
// WMMA subtiles -> 16 v_wmma per k-step, each frag reused 4x).
// Double-buffered LDS (40KB): global loads for tile k+1 issue before the
// 16 WMMAs on tile k; convert + ds_store after; one barrier per iteration.
// ---------------------------------------------------------------------------
__global__ __launch_bounds__(128) void fp4_gemm_kernel(
    const float* __restrict__ X, const _Float16* __restrict__ W,
    const float* __restrict__ bias, float* __restrict__ out)
{
    __shared__ _Float16 As[2][BM * SLDS];
    __shared__ _Float16 Bs[2][BN * SLDS];

    const int tid  = threadIdx.x;
    const int wave = tid >> 5;
    const int lane = tid & 31;
    const int wm   = (wave & 1) * 64;     // wave row strip
    const int wn   = (wave >> 1) * 64;    // wave col strip
    const int m0   = blockIdx.y * BM;
    const int n0   = blockIdx.x * BN;

    const int lrow = lane & 15;           // index within 16x16 tile
    const int lk   = (lane >> 4) << 3;    // lanes 0-15: K base 0; 16-31: K base 8

    v8f acc[4][4];
#pragma unroll
    for (int mt = 0; mt < 4; ++mt)
#pragma unroll
        for (int nt = 0; nt < 4; ++nt)
#pragma unroll
            for (int r = 0; r < 8; ++r) acc[mt][nt][r] = 0.0f;

    // each thread stages one full row slice per tile: 32 f32 of x, 32 f16 of W
    const float*    xsrc = X + (size_t)(m0 + tid) * IN_F;
    const _Float16* wsrc = W + (size_t)(n0 + tid) * IN_F;

    // ---- prologue: stage tile k0=0 into buffer 0 ----
    {
#pragma unroll
        for (int i = 0; i < 4; ++i) {
            float4 f0 = *(const float4*)(xsrc + i * 8);
            float4 f1 = *(const float4*)(xsrc + i * 8 + 4);
            v8h h;
            h[0] = (_Float16)f0.x; h[1] = (_Float16)f0.y;
            h[2] = (_Float16)f0.z; h[3] = (_Float16)f0.w;
            h[4] = (_Float16)f1.x; h[5] = (_Float16)f1.y;
            h[6] = (_Float16)f1.z; h[7] = (_Float16)f1.w;
            *(v8h*)(&As[0][tid * SLDS + i * 8]) = h;
            *(v8h*)(&Bs[0][tid * SLDS + i * 8]) = *(const v8h*)(wsrc + i * 8);
        }
    }
    __syncthreads();

    for (int k0 = 0; k0 < IN_F; k0 += BK) {
        const int  cur  = (k0 >> 5) & 1;
        const bool more = (k0 + BK) < IN_F;   // scalar-uniform branch

        // ---- issue next tile's global loads first (latency overlap) ----
        float4 fa[8];
        v8h    hb[4];
        if (more) {
            const float*    xs = xsrc + k0 + BK;
            const _Float16* ws = wsrc + k0 + BK;
#pragma unroll
            for (int i = 0; i < 8; ++i) fa[i] = *(const float4*)(xs + i * 4);
#pragma unroll
            for (int i = 0; i < 4; ++i) hb[i] = *(const v8h*)(ws + i * 8);
        }

        // ---- compute on current buffer: 16 frag loads, 16 WMMAs ----
        v16h a[4], b[4];
#pragma unroll
        for (int t = 0; t < 4; ++t) {
            const _Float16* pa = &As[cur][(wm + t * 16 + lrow) * SLDS + lk];
            v8h alo = *(const v8h*)pa;
            v8h ahi = *(const v8h*)(pa + 16);
            a[t] = __builtin_shufflevector(alo, ahi,
                   0,1,2,3,4,5,6,7,8,9,10,11,12,13,14,15);
            const _Float16* pb = &Bs[cur][(wn + t * 16 + lrow) * SLDS + lk];
            v8h blo = *(const v8h*)pb;
            v8h bhi = *(const v8h*)(pb + 16);
            b[t] = __builtin_shufflevector(blo, bhi,
                   0,1,2,3,4,5,6,7,8,9,10,11,12,13,14,15);
        }
#pragma unroll
        for (int mt = 0; mt < 4; ++mt)
#pragma unroll
            for (int nt = 0; nt < 4; ++nt)
                acc[mt][nt] = __builtin_amdgcn_wmma_f32_16x16x32_f16(
                    false, a[mt], false, b[nt],
                    (short)0, acc[mt][nt], false, false);

        // ---- convert + store next tile into other buffer ----
        if (more) {
            const int nxt = cur ^ 1;
#pragma unroll
            for (int i = 0; i < 4; ++i) {
                v8h h;
                h[0] = (_Float16)fa[2*i].x;   h[1] = (_Float16)fa[2*i].y;
                h[2] = (_Float16)fa[2*i].z;   h[3] = (_Float16)fa[2*i].w;
                h[4] = (_Float16)fa[2*i+1].x; h[5] = (_Float16)fa[2*i+1].y;
                h[6] = (_Float16)fa[2*i+1].z; h[7] = (_Float16)fa[2*i+1].w;
                *(v8h*)(&As[nxt][tid * SLDS + i * 8]) = h;
                *(v8h*)(&Bs[nxt][tid * SLDS + i * 8]) = hb[i];
            }
        }
        __syncthreads();
    }

    // ---- epilogue: C layout VGPR r -> M = r + (lane>=16 ? 8 : 0), N = lane&15 ----
    const int erow = (lane >> 4) << 3;
#pragma unroll
    for (int mt = 0; mt < 4; ++mt) {
#pragma unroll
        for (int nt = 0; nt < 4; ++nt) {
            const int col = n0 + wn + nt * 16 + lrow;
            const float bv = bias[col];
            size_t base = (size_t)(m0 + wm + mt * 16 + erow) * (size_t)OUT_F + col;
#pragma unroll
            for (int r = 0; r < 8; ++r)
                out[base + (size_t)r * OUT_F] = acc[mt][nt][r] + bv;
        }
    }
}

// ---------------------------------------------------------------------------
extern "C" void kernel_launch(void* const* d_in, const int* in_sizes, int n_in,
                              void* d_out, int out_size, void* d_ws, size_t ws_size,
                              hipStream_t stream)
{
    const float* x      = (const float*)d_in[0];
    const int*   wp     = (const int*)d_in[1];
    const float* scales = (const float*)d_in[2];
    const float* bias   = (const float*)d_in[3];
    float*       out    = (float*)d_out;
    _Float16*    Wh     = (_Float16*)d_ws;   // 32 MB f16 weight matrix

    const int n_packed = in_sizes[1];               // OUT_F*IN_F/2
    const int M        = in_sizes[0] / IN_F;        // 8192

    dequant_fp4_kernel<<<(n_packed + 255) / 256, 256, 0, stream>>>(wp, scales, Wh, n_packed);

    dim3 grid(OUT_F / BN, M / BM);
    fp4_gemm_kernel<<<grid, 128, 0, stream>>>(x, Wh, bias, out);
}
